// Attention_Block_defAttentionSequSize_19413252177932
// MI455X (gfx1250) — compile-verified, matrix operand order checked by reading, not run on hardware
//
#include <hip/hip_runtime.h>

// ---------------------------------------------------------------------------
// MI455X (gfx1250) implementation of the blocked-attention reference.
// All GEMMs run on v_wmma_f32_16x16x32_bf16 (wave32, 16x16 tiles, f32 acc).
// Compute-bound problem (~66 GFLOP vs ~50MB traffic -> ~1300 FLOP/byte),
// so matrix-core utilization dominates; bf16 keeps fp32 dynamic range.
// LDS staging: double-buffered GLOBAL_LOAD_ASYNC_TO_LDS_B128 (ASYNCcnt) so the
// global->LDS DMA for K-step i+1 overlaps the WMMAs of K-step i.
// ---------------------------------------------------------------------------

typedef __attribute__((ext_vector_type(16))) __bf16          v16bf;
typedef __attribute__((ext_vector_type(8)))  float           v8f;
typedef __attribute__((ext_vector_type(8)))  unsigned short  v8u;
typedef __attribute__((ext_vector_type(4)))  int             v4i;

#if defined(__has_builtin)
#  if __has_builtin(__builtin_amdgcn_global_load_async_to_lds_b128) && \
      __has_builtin(__builtin_amdgcn_s_wait_asynccnt)
#    define USE_ASYNC_LDS 1
#  endif
#endif
#ifndef USE_ASYNC_LDS
#  define USE_ASYNC_LDS 0
#endif

#if USE_ASYNC_LDS
// Direct global->LDS async copy, 16B per lane, tracked by ASYNCcnt.
__device__ __forceinline__ void async_cp16(const unsigned short* g, unsigned short* l) {
  __builtin_amdgcn_global_load_async_to_lds_b128(
      (__attribute__((address_space(1))) v4i*)g,
      (__attribute__((address_space(3))) v4i*)l, 0, 0);
}
#endif

__device__ __forceinline__ unsigned short f2bf(float f) {
  unsigned int u = __builtin_bit_cast(unsigned int, f);
  u += 0x7fffu + ((u >> 16) & 1u);            // round-to-nearest-even
  return (unsigned short)(u >> 16);
}

__device__ __forceinline__ v16bf mkop(v8u lo, v8u hi) {
  union { v8u h[2]; v16bf v; } u;
  u.h[0] = lo; u.h[1] = hi;
  return u.v;
}
// Build a WMMA 16-bit operand: two contiguous 8-element (16B) chunks.
// ISA 16-bit A 16x32 layout: lane-half 0 -> elems0..7 = K0..7, elems8..15 = K16..23
//                            lane-half 1 -> elems0..7 = K8..15, elems8..15 = K24..31
__device__ __forceinline__ v16bf ld2(const unsigned short* p0, const unsigned short* p1) {
  return mkop(*(const v8u*)p0, *(const v8u*)p1);
}

// ---------------------------------------------------------------------------
// Small data-movement kernels
// ---------------------------------------------------------------------------
__global__ void f32_to_bf16_k(const float* __restrict__ x, unsigned short* __restrict__ y, int n) {
  int i = blockIdx.x * 256 + threadIdx.x;
  if (i < n) y[i] = f2bf(x[i]);
}

// W (K x N, row-major f32) -> Wt (N x K, row-major bf16)
__global__ void transposeW_k(const float* __restrict__ W, unsigned short* __restrict__ Wt,
                             int K, int N) {
  int i = blockIdx.x * 256 + threadIdx.x;
  if (i < N * K) {
    int nrow = i / K, k = i - nrow * K;
    Wt[i] = f2bf(W[(size_t)k * N + nrow]);
  }
}

// Residual row map for the Wo epilogue: projected row R -> raw value row
__global__ void build_resmap_k(int* __restrict__ m) {
  int R = blockIdx.x * 256 + threadIdx.x;
  if (R < 15360) {
    int blkrow = R >> 7, s = R & 127;
    int batch = blkrow / 15, blk = blkrow - batch * 15;
    m[R] = batch * 1024 + blk * 64 + s;
  }
}

// ---------------------------------------------------------------------------
// Generic bf16 WMMA GEMM:  C(MxN,f32) = A(MxK) * Bt(NxK)^T + bias + resid
// 128x128 tile per workgroup, 8 waves x (16 rows x 128 cols), K-step 32.
// Double-buffered async LDS staging: DMA of step i+1 overlaps WMMAs of step i.
// ---------------------------------------------------------------------------
__global__ __launch_bounds__(256)
void gemm_bf16_k(const unsigned short* __restrict__ A,
                 const unsigned short* __restrict__ Bt,
                 const float* __restrict__ bias,
                 const float* __restrict__ resid,
                 const int*   __restrict__ rmap,
                 float* __restrict__ outF,
                 unsigned short* __restrict__ outB,
                 int M, int N, int K) {
  __shared__ alignas(16) unsigned short As[2][128 * 32];
  __shared__ alignas(16) unsigned short Bs[2][128 * 32];
  const int row0 = blockIdx.y * 128;
  const int col0 = blockIdx.x * 128;
  const int tid  = threadIdx.x;
  const int w    = tid >> 5;
  const int l    = tid & 31;
  const int n16  = l & 15;
  const int hl   = l >> 4;
  const int srow  = tid >> 1;          // staging: 2 threads per row
  const int shalf = (tid & 1) * 16;    // staging: 16-col half

  v8f zero8 = {0.f, 0.f, 0.f, 0.f, 0.f, 0.f, 0.f, 0.f};
  v8f acc[8];
#pragma unroll
  for (int i = 0; i < 8; ++i) acc[i] = zero8;

  const unsigned short* aRow = A  + (size_t)(row0 + srow) * K + shalf;
  const unsigned short* bRow = Bt + (size_t)(col0 + srow) * K + shalf;
  const int sdst = srow * 32 + shalf;

  auto stage = [&](int buf, int k0) {
    const unsigned short* ap = aRow + k0;
    const unsigned short* bp = bRow + k0;
    __builtin_prefetch(ap + 64, 0, 3);   // two slabs ahead, keep near
    __builtin_prefetch(bp + 64, 0, 3);
#if USE_ASYNC_LDS
    async_cp16(ap,     &As[buf][sdst]);
    async_cp16(ap + 8, &As[buf][sdst + 8]);
    async_cp16(bp,     &Bs[buf][sdst]);
    async_cp16(bp + 8, &Bs[buf][sdst + 8]);
#else
    *(v8u*)&As[buf][sdst]     = *(const v8u*)ap;
    *(v8u*)&As[buf][sdst + 8] = *(const v8u*)(ap + 8);
    *(v8u*)&Bs[buf][sdst]     = *(const v8u*)bp;
    *(v8u*)&Bs[buf][sdst + 8] = *(const v8u*)(bp + 8);
#endif
  };

  const int nsteps = K >> 5;
  stage(0, 0);
  int cur = 0;
  for (int i = 0; i < nsteps; ++i) {
#if USE_ASYNC_LDS
    __builtin_amdgcn_s_wait_asynccnt(0);   // this wave's DMA into buf[cur] done
#endif
    __syncthreads();                        // all waves staged cur; prior reads of cur^1 done
    if (i + 1 < nsteps) stage(cur ^ 1, (i + 1) * 32);  // background DMA overlaps WMMAs

    // Preload all 8 B operands, then stream 8 back-to-back WMMAs
    v16bf aop = ld2(&As[cur][(w * 16 + n16) * 32 + hl * 8],
                    &As[cur][(w * 16 + n16) * 32 + 16 + hl * 8]);
    v16bf bop[8];
#pragma unroll
    for (int nt = 0; nt < 8; ++nt)
      bop[nt] = ld2(&Bs[cur][(nt * 16 + n16) * 32 + hl * 8],
                    &Bs[cur][(nt * 16 + n16) * 32 + 16 + hl * 8]);
#pragma unroll
    for (int nt = 0; nt < 8; ++nt)
      acc[nt] = __builtin_amdgcn_wmma_f32_16x16x32_bf16(
          false, aop, false, bop[nt], (short)0, acc[nt], false, false);
    cur ^= 1;
  }

#pragma unroll
  for (int nt = 0; nt < 8; ++nt) {
#pragma unroll
    for (int r = 0; r < 8; ++r) {
      int grow = row0 + w * 16 + hl * 8 + r;   // C layout: VGPR r -> M=r / M=r+8
      int gcol = col0 + nt * 16 + n16;         // N = lane&15
      float v = acc[nt][r];
      if (bias)  v += bias[gcol];
      if (resid) {
        int rr = rmap ? rmap[grow] : grow;
        v += resid[(size_t)rr * N + gcol];
      }
      if (outF) outF[(size_t)grow * N + gcol] = v;
      if (outB) outB[(size_t)grow * N + gcol] = f2bf(v);
    }
  }
}

// ---------------------------------------------------------------------------
// Attention: one workgroup per (block b in [0,120), head h in [0,8)).
// Handles the reference's head-mixing reshape via the row/col index map:
//   q_h[s,d] = P[batch*1024 + blk*64 + h*16 + s/8][(s%8)*256 + d]
// ---------------------------------------------------------------------------
__global__ __launch_bounds__(256)
void attn_k(const unsigned short* __restrict__ Pq,
            const unsigned short* __restrict__ Pk,
            const unsigned short* __restrict__ Pv,
            unsigned short* __restrict__ vals) {
  __shared__ alignas(16) unsigned short ps [128 * 128];  // softmax probs (bf16)
  __shared__ alignas(16) unsigned short vts[128 * 128];  // v^T half-tile (bf16)
  const int b = blockIdx.x;       // 0..119
  const int h = blockIdx.y;       // 0..7
  const int batch = b / 15, blk = b - batch * 15;
  const int rowbase = batch * 1024 + blk * 64 + h * 16;
  const int tid = threadIdx.x;
  const int w   = tid >> 5;
  const int l   = tid & 31;
  const int n16 = l & 15;
  const int hl  = l >> 4;

  // ---- scores = q k^T / 16, then row softmax ----
  const int sA = w * 16 + n16;    // this lane's q row (0..127)
  const unsigned short* aptr =
      Pq + (size_t)(rowbase + (sA >> 3)) * 2048 + (sA & 7) * 256;
  const unsigned short* bptr[8];
#pragma unroll
  for (int nt = 0; nt < 8; ++nt) {
    int t = nt * 16 + n16;        // this lane's k row for column-tile nt
    bptr[nt] = Pk + (size_t)(rowbase + (t >> 3)) * 2048 + (t & 7) * 256;
  }

  v8f zero8 = {0.f, 0.f, 0.f, 0.f, 0.f, 0.f, 0.f, 0.f};
  v8f acc[8];
#pragma unroll
  for (int i = 0; i < 8; ++i) acc[i] = zero8;

#pragma unroll
  for (int kk = 0; kk < 8; ++kk) {
    int d0 = kk * 32 + hl * 8;
    v16bf aop = ld2(aptr + d0, aptr + d0 + 16);
    v16bf bop[8];
#pragma unroll
    for (int nt = 0; nt < 8; ++nt)
      bop[nt] = ld2(bptr[nt] + d0, bptr[nt] + d0 + 16);
#pragma unroll
    for (int nt = 0; nt < 8; ++nt)
      acc[nt] = __builtin_amdgcn_wmma_f32_16x16x32_bf16(
          false, aop, false, bop[nt], (short)0, acc[nt], false, false);
  }

  const float scale = 0.0625f;    // 1/sqrt(256)
#pragma unroll
  for (int r = 0; r < 8; ++r) {   // row (w*16 + hl*8 + r): 128 cols across 8 tiles x 16 lanes
    float mx = -3.0e30f;
#pragma unroll
    for (int nt = 0; nt < 8; ++nt) mx = fmaxf(mx, acc[nt][r]);
    mx = fmaxf(mx, __shfl_xor(mx, 1, 16));
    mx = fmaxf(mx, __shfl_xor(mx, 2, 16));
    mx = fmaxf(mx, __shfl_xor(mx, 4, 16));
    mx = fmaxf(mx, __shfl_xor(mx, 8, 16));
    float sum = 0.f;
#pragma unroll
    for (int nt = 0; nt < 8; ++nt) {
      float e = __expf((acc[nt][r] - mx) * scale);
      acc[nt][r] = e;
      sum += e;
    }
    sum += __shfl_xor(sum, 1, 16);
    sum += __shfl_xor(sum, 2, 16);
    sum += __shfl_xor(sum, 4, 16);
    sum += __shfl_xor(sum, 8, 16);
    float inv = 1.0f / sum;
    int prow = w * 16 + hl * 8 + r;
#pragma unroll
    for (int nt = 0; nt < 8; ++nt)
      ps[prow * 128 + nt * 16 + n16] = f2bf(acc[nt][r] * inv);
  }

  // ---- out = p @ v, two 128-wide d-halves (v^T staged in LDS) ----
  size_t obase = (size_t)b * 262144 + (size_t)h * 32768;
  for (int dh = 0; dh < 2; ++dh) {
    __syncthreads();
    {
      int t     = tid >> 1;               // v row 0..127
      int dpart = (tid & 1) * 64;         // d sub-half
      const unsigned short* vp =
          Pv + (size_t)(rowbase + (t >> 3)) * 2048 + (t & 7) * 256 + dh * 128 + dpart;
#pragma unroll
      for (int j = 0; j < 64; j += 8) {
        v8u x = *(const v8u*)(vp + j);
#pragma unroll
        for (int e = 0; e < 8; ++e)
          vts[(dpart + j + e) * 128 + t] = x[e];   // transpose into LDS
      }
    }
    __syncthreads();

    v8f acc2[8];
#pragma unroll
    for (int i = 0; i < 8; ++i) acc2[i] = zero8;
#pragma unroll
    for (int kk = 0; kk < 4; ++kk) {
      int k0 = kk * 32 + hl * 8;
      v16bf aop = ld2(&ps[sA * 128 + k0], &ps[sA * 128 + k0 + 16]);
      v16bf bop[8];
#pragma unroll
      for (int nt = 0; nt < 8; ++nt)
        bop[nt] = ld2(&vts[(nt * 16 + n16) * 128 + k0],
                      &vts[(nt * 16 + n16) * 128 + k0 + 16]);
#pragma unroll
      for (int nt = 0; nt < 8; ++nt)
        acc2[nt] = __builtin_amdgcn_wmma_f32_16x16x32_bf16(
            false, aop, false, bop[nt], (short)0, acc2[nt], false, false);
    }
#pragma unroll
    for (int nt = 0; nt < 8; ++nt) {
#pragma unroll
      for (int r = 0; r < 8; ++r) {
        int s = w * 16 + hl * 8 + r;
        int d = dh * 128 + nt * 16 + n16;
        vals[obase + (size_t)s * 256 + d] = f2bf(acc2[nt][r]);
      }
    }
  }
}

// ---------------------------------------------------------------------------
// LayerNorm over 256-wide rows: one wave32 per row (8 rows per block).
// ---------------------------------------------------------------------------
__global__ __launch_bounds__(256)
void ln_rows_k(const float* __restrict__ X, const float* __restrict__ g,
               const float* __restrict__ bta, float* __restrict__ outF,
               unsigned short* __restrict__ outB, int rows) {
  int w = threadIdx.x >> 5, l = threadIdx.x & 31;
  int row = blockIdx.x * 8 + w;
  if (row >= rows) return;
  const float* xr = X + (size_t)row * 256 + l * 8;
  float x[8];
  *(float4*)&x[0] = *(const float4*)xr;
  *(float4*)&x[4] = *(const float4*)(xr + 4);
  float s = 0.f;
#pragma unroll
  for (int j = 0; j < 8; ++j) s += x[j];
#pragma unroll
  for (int off = 16; off >= 1; off >>= 1) s += __shfl_xor(s, off, 32);
  float mu = s * (1.0f / 256.0f);
  float vs = 0.f;
#pragma unroll
  for (int j = 0; j < 8; ++j) { float d = x[j] - mu; vs += d * d; }
#pragma unroll
  for (int off = 16; off >= 1; off >>= 1) vs += __shfl_xor(vs, off, 32);
  float rs = rsqrtf(vs * (1.0f / 256.0f) + 1e-5f);
  int c0 = l * 8;
  float o[8];
  v8u ob;
#pragma unroll
  for (int j = 0; j < 8; ++j) {
    o[j] = (x[j] - mu) * rs * g[c0 + j] + bta[c0 + j];
    ob[j] = f2bf(o[j]);
  }
  if (outF) {
    *(float4*)(outF + (size_t)row * 256 + c0)     = *(const float4*)&o[0];
    *(float4*)(outF + (size_t)row * 256 + c0 + 4) = *(const float4*)&o[4];
  }
  if (outB) *(v8u*)(outB + (size_t)row * 256 + c0) = ob;
}

// LN2 fused with the de-overlap gather into the final output layout.
__global__ __launch_bounds__(256)
void ln2_scatter_k(const float* __restrict__ X2, const float* __restrict__ g,
                   const float* __restrict__ bta, float* __restrict__ out) {
  int w = threadIdx.x >> 5, l = threadIdx.x & 31;
  int orow = blockIdx.x * 8 + w;          // 0..8191 output rows
  int batch = orow >> 10, t = orow & 1023;
  int blk, s;
  if (t < 96)        { blk = 0;  s = t; }
  else if (t >= 928) { blk = 14; s = t - 896; }
  else               { blk = 1 + (t - 96) / 64; s = 32 + ((t - 96) & 63); }
  int R = (batch * 15 + blk) * 128 + s;
  const float* xr = X2 + (size_t)R * 256 + l * 8;
  float x[8];
  *(float4*)&x[0] = *(const float4*)xr;
  *(float4*)&x[4] = *(const float4*)(xr + 4);
  float sum = 0.f;
#pragma unroll
  for (int j = 0; j < 8; ++j) sum += x[j];
#pragma unroll
  for (int off = 16; off >= 1; off >>= 1) sum += __shfl_xor(sum, off, 32);
  float mu = sum * (1.0f / 256.0f);
  float vs = 0.f;
#pragma unroll
  for (int j = 0; j < 8; ++j) { float d = x[j] - mu; vs += d * d; }
#pragma unroll
  for (int off = 16; off >= 1; off >>= 1) vs += __shfl_xor(vs, off, 32);
  float rs = rsqrtf(vs * (1.0f / 256.0f) + 1e-5f);
  int c0 = l * 8;
  float o[8];
#pragma unroll
  for (int j = 0; j < 8; ++j)
    o[j] = (x[j] - mu) * rs * g[c0 + j] + bta[c0 + j];
  *(float4*)(out + (size_t)orow * 256 + c0)     = *(const float4*)&o[0];
  *(float4*)(out + (size_t)orow * 256 + c0 + 4) = *(const float4*)&o[4];
}

// ---------------------------------------------------------------------------
// Host-side orchestration
// ---------------------------------------------------------------------------
extern "C" void kernel_launch(void* const* d_in, const int* in_sizes, int n_in,
                              void* d_out, int out_size, void* d_ws, size_t ws_size,
                              hipStream_t stream) {
  (void)in_sizes; (void)n_in; (void)out_size; (void)ws_size;
  const float* query = (const float*)d_in[0];
  const float* key   = (const float*)d_in[1];
  const float* value = (const float*)d_in[2];
  const float* Wq = (const float*)d_in[3];   const float* bq = (const float*)d_in[4];
  const float* Wk = (const float*)d_in[5];   const float* bk = (const float*)d_in[6];
  const float* Wv = (const float*)d_in[7];   const float* bv = (const float*)d_in[8];
  const float* Wo = (const float*)d_in[9];   const float* bo = (const float*)d_in[10];
  const float* ln1g = (const float*)d_in[11]; const float* ln1b = (const float*)d_in[12];
  const float* Wff  = (const float*)d_in[13]; const float* bff  = (const float*)d_in[14];
  const float* ln2g = (const float*)d_in[15]; const float* ln2b = (const float*)d_in[16];

  const size_t SR = 8192, Dm = 256, HD = 2048, MR = 15360;

  char* p = (char*)d_ws;
  auto alloc = [&](size_t bytes) -> char* {
    char* r = p;
    p += (bytes + 255) & ~(size_t)255;
    return r;
  };
  unsigned short* xq   = (unsigned short*)alloc(SR * Dm * 2);
  unsigned short* xk   = (unsigned short*)alloc(SR * Dm * 2);
  unsigned short* xv   = (unsigned short*)alloc(SR * Dm * 2);
  unsigned short* wqt  = (unsigned short*)alloc(HD * Dm * 2);
  unsigned short* wkt  = (unsigned short*)alloc(HD * Dm * 2);
  unsigned short* wvt  = (unsigned short*)alloc(HD * Dm * 2);
  unsigned short* wot  = (unsigned short*)alloc(Dm * HD * 2);
  unsigned short* wfft = (unsigned short*)alloc(Dm * Dm * 2);
  unsigned short* pq   = (unsigned short*)alloc(SR * HD * 2);
  unsigned short* pk   = (unsigned short*)alloc(SR * HD * 2);
  unsigned short* pv   = (unsigned short*)alloc(SR * HD * 2);
  unsigned short* vals = (unsigned short*)alloc(MR * HD * 2);
  float*          x1   = (float*)alloc(MR * Dm * 4);
  float*          x1n  = (float*)alloc(MR * Dm * 4);
  unsigned short* x1nb = (unsigned short*)alloc(MR * Dm * 2);
  float*          x2   = (float*)alloc(MR * Dm * 4);
  int*            rmap = (int*)alloc(MR * 4);

  const dim3 blkT(256);

  // 1) precision conversions (bandwidth-trivial)
  int nIn = (int)(SR * Dm);
  f32_to_bf16_k<<<nIn / 256, blkT, 0, stream>>>(query, xq, nIn);
  f32_to_bf16_k<<<nIn / 256, blkT, 0, stream>>>(key,   xk, nIn);
  f32_to_bf16_k<<<nIn / 256, blkT, 0, stream>>>(value, xv, nIn);
  transposeW_k<<<(int)(HD * Dm) / 256, blkT, 0, stream>>>(Wq,  wqt, 256, 2048);
  transposeW_k<<<(int)(HD * Dm) / 256, blkT, 0, stream>>>(Wk,  wkt, 256, 2048);
  transposeW_k<<<(int)(HD * Dm) / 256, blkT, 0, stream>>>(Wv,  wvt, 256, 2048);
  transposeW_k<<<(int)(Dm * HD) / 256, blkT, 0, stream>>>(Wo,  wot, 2048, 256);
  transposeW_k<<<(int)(Dm * Dm) / 256, blkT, 0, stream>>>(Wff, wfft, 256, 256);
  build_resmap_k<<<60, blkT, 0, stream>>>(rmap);

  // 2) QKV projections over the full (un-gathered) sequence: 8192x2048 = X @ W + b
  gemm_bf16_k<<<dim3(16, 64), blkT, 0, stream>>>(xq, wqt, bq, nullptr, nullptr,
                                                 nullptr, pq, 8192, 2048, 256);
  gemm_bf16_k<<<dim3(16, 64), blkT, 0, stream>>>(xk, wkt, bk, nullptr, nullptr,
                                                 nullptr, pk, 8192, 2048, 256);
  gemm_bf16_k<<<dim3(16, 64), blkT, 0, stream>>>(xv, wvt, bv, nullptr, nullptr,
                                                 nullptr, pv, 8192, 2048, 256);

  // 3) blocked attention (gather + softmax + A@V), vals stored in (blk,h,s,d) flat
  attn_k<<<dim3(120, 8), blkT, 0, stream>>>(pq, pk, pv, vals);

  // 4) output projection + vb residual (gathered via rmap) -> x1 (f32)
  gemm_bf16_k<<<dim3(2, 120), blkT, 0, stream>>>(vals, wot, bo, value, rmap,
                                                 x1, nullptr, 15360, 256, 2048);

  // 5) LN1 -> f32 (residual for FF) + bf16 (GEMM A operand)
  ln_rows_k<<<1920, blkT, 0, stream>>>(x1, ln1g, ln1b, x1n, x1nb, 15360);

  // 6) FF + identity residual -> x2 (f32)
  gemm_bf16_k<<<dim3(2, 120), blkT, 0, stream>>>(x1nb, wfft, bff, x1n, nullptr,
                                                 x2, nullptr, 15360, 256, 256);

  // 7) LN2 + de-overlap gather into final (8,1024,256) f32 output
  ln2_scatter_k<<<1024, blkT, 0, stream>>>(x2, ln2g, ln2b, (float*)d_out);
}
